// KoLeoLoss_6794638262393
// MI455X (gfx1250) — matrix operand order repaired and checked by reading, not compiled, not run on hardware
//
#include <hip/hip_runtime.h>

#define N_ROWS 8192
#define DIM    384
#define KSTEPS 12          // 384 / 32
#define NTILES (N_ROWS / 32)
#define EPSF   1e-8f

typedef __attribute__((ext_vector_type(16))) _Float16 v16h;
typedef __attribute__((ext_vector_type(8)))  float    v8f;

union F16Frag { v16h h; uint4 u[2]; };

// --- CDNA5 async global->LDS path (ASYNCcnt-tracked), with fallback --------
#if defined(__has_builtin)
#if __has_builtin(__builtin_amdgcn_global_load_async_to_lds_b128)
#define USE_ASYNC_LDS 1
#endif
#if __has_builtin(__builtin_amdgcn_s_wait_asynccnt)
#define WAIT_ASYNC() __builtin_amdgcn_s_wait_asynccnt(0)
#endif
#endif
#ifndef USE_ASYNC_LDS
#define USE_ASYNC_LDS 0
#endif
#ifndef WAIT_ASYNC
#define WAIT_ASYNC() asm volatile("s_wait_asynccnt 0" ::: "memory")
#endif

// Builtin signature (from hipcc diagnostic): param0 is
//   'int __vector(4) __attribute__((address_space(1)))*'  (AS1 printed as __device__)
typedef int v4i_gcc __attribute__((__vector_size__(4 * sizeof(int))));
typedef __attribute__((address_space(1))) v4i_gcc* as1_v4i_p;
typedef __attribute__((address_space(3))) v4i_gcc* as3_v4i_p;

// ---------------------------------------------------------------------------
// Phase 1: row L2-normalize; emit f32 copy (exact distance phase) and f16
// copy (WMMA operands). One wave per row, coalesced stride-32 access.
// ---------------------------------------------------------------------------
__global__ void __launch_bounds__(256)
koleo_normalize(const float* __restrict__ in,
                float* __restrict__ xf,
                unsigned short* __restrict__ xh) {
  const int lane = threadIdx.x & 31;
  const int row  = blockIdx.x * 8 + (threadIdx.x >> 5);
  const float* r = in + (size_t)row * DIM;
  float v[12];
  float s = 0.f;
#pragma unroll
  for (int i = 0; i < 12; ++i) { v[i] = r[lane + i * 32]; s += v[i] * v[i]; }
#pragma unroll
  for (int m = 16; m >= 1; m >>= 1) s += __shfl_xor(s, m, 32);
  const float rn = 1.0f / fmaxf(sqrtf(s), EPSF);

  float* of = xf + (size_t)row * DIM;
  unsigned short* oh = xh + (size_t)row * DIM;
#pragma unroll
  for (int i = 0; i < 12; ++i) {
    float x = v[i] * rn;
    of[lane + i * 32] = x;
    _Float16 hx = (_Float16)x;
    oh[lane + i * 32] = __builtin_bit_cast(unsigned short, hx);
  }
}

// ---------------------------------------------------------------------------
// Phase 2: fused X·X^T + diagonal mask + row argmax via WMMA, with a
// double-buffered LDS pipeline: async global->LDS copies of tile j+1 are in
// flight while tile j is consumed by v_wmma_f32_16x16x32_f16.
// 128 blocks x 4 waves; wave owns rows [rowBase, rowBase+16); A fragments
// (16 rows x 384 K) hoisted to registers per the CDNA5 16-bit A layout.
// ---------------------------------------------------------------------------
__global__ void __launch_bounds__(128)
koleo_argmax(const unsigned short* __restrict__ xh, int* __restrict__ nn) {
  __shared__ unsigned short lb[2][32 * DIM];   // 2 x 24 KB candidate stage
  const int tid   = threadIdx.x;
  const int lane  = tid & 31;
  const int wave  = tid >> 5;
  const int lsub  = lane & 15;
  const int halfv = lane >> 4;
  const int rowBase = blockIdx.x * 64 + wave * 16;

  // Hoist A fragments for this wave's 16 rows (x_f16 read once per wave).
  F16Frag afrag[KSTEPS];
  const int arow = rowBase + lsub;
#pragma unroll
  for (int kk = 0; kk < KSTEPS; ++kk) {
    const uint4* p = (const uint4*)(xh + (size_t)arow * DIM + kk * 32 + halfv * 8);
    afrag[kk].u[0] = p[0];   // K = kk*32 + halfv*8      .. +7
    afrag[kk].u[1] = p[2];   // K = kk*32 + 16 + halfv*8 .. +7
  }

  float bestv[8];
  int   besti[8];
#pragma unroll
  for (int r = 0; r < 8; ++r) { bestv[r] = -3.4e38f; besti[r] = 0; }

#if USE_ASYNC_LDS
  // ---- async double-buffered pipeline ----
  {
    const char* g0 = (const char*)xh;                 // tile 0 at offset 0
#pragma unroll
    for (int i = 0; i < 12; ++i) {
      const size_t off = (size_t)(tid + i * 128) * 16;
      __builtin_amdgcn_global_load_async_to_lds_b128(
          (as1_v4i_p)(uintptr_t)(g0 + off),
          (as3_v4i_p)(uintptr_t)((char*)lb[0] + off), 0, 0);
    }
  }
  WAIT_ASYNC();
  __syncthreads();

  int cur = 0;
  for (int j = 0; j < NTILES; ++j) {
    if (j + 1 < NTILES) {                             // prefetch tile j+1
      const char* gs = (const char*)(xh + (size_t)(j + 1) * 32 * DIM);
#pragma unroll
      for (int i = 0; i < 12; ++i) {
        const size_t off = (size_t)(tid + i * 128) * 16;
        __builtin_amdgcn_global_load_async_to_lds_b128(
            (as1_v4i_p)(uintptr_t)(gs + off),
            (as3_v4i_p)(uintptr_t)((char*)lb[cur ^ 1] + off), 0, 0);
      }
    }
#else
  // ---- fallback: software-pipelined register round-trip ----
  uint4 stage[12];
  {
    const uint4* src = (const uint4*)xh;
    uint4* dst = (uint4*)lb[0];
#pragma unroll
    for (int i = 0; i < 12; ++i) stage[i] = src[tid + i * 128];
#pragma unroll
    for (int i = 0; i < 12; ++i) dst[tid + i * 128] = stage[i];
  }
  __syncthreads();

  int cur = 0;
  for (int j = 0; j < NTILES; ++j) {
    if (j + 1 < NTILES) {                             // issue loads for j+1
      const uint4* src = (const uint4*)(xh + (size_t)(j + 1) * 32 * DIM);
#pragma unroll
      for (int i = 0; i < 12; ++i) stage[i] = src[tid + i * 128];
    }
#endif
    // ---- compute tile j from lb[cur] ----
    const unsigned short* lbc = lb[cur];
    const int jBase = j * 32;
#pragma unroll
    for (int t = 0; t < 2; ++t) {
      v8f c = {0.f, 0.f, 0.f, 0.f, 0.f, 0.f, 0.f, 0.f};
#pragma unroll
      for (int kk = 0; kk < KSTEPS; ++kk) {
        F16Frag b;
        const uint4* lp =
            (const uint4*)(lbc + ((t * 16 + lsub) * DIM + kk * 32 + halfv * 8));
        b.u[0] = lp[0];   // K = kk*32 + halfv*8      .. +7
        b.u[1] = lp[2];   // K = kk*32 + 16 + halfv*8 .. +7
        c = __builtin_amdgcn_wmma_f32_16x16x32_f16(
                false, afrag[kk].h, false, b.h, (short)0, c, false, false);
      }
      const int col = jBase + t * 16 + lsub;
#pragma unroll
      for (int r = 0; r < 8; ++r) {
        float vv = c[r];
        const int row = rowBase + halfv * 8 + r;
        if (row == col) vv = -1.0f;                 // diagonal mask
        if (vv > bestv[r]) { bestv[r] = vv; besti[r] = col; }  // keep earliest
      }
    }

#if USE_ASYNC_LDS
    WAIT_ASYNC();          // this wave's copies into lb[cur^1] landed
    __syncthreads();       // all waves: copies visible, reads of lb[cur] done
#else
    if (j + 1 < NTILES) {
      uint4* dst = (uint4*)lb[cur ^ 1];
#pragma unroll
      for (int i = 0; i < 12; ++i) dst[tid + i * 128] = stage[i];
    }
    __syncthreads();
#endif
    cur ^= 1;
  }

  // Reduce (max, earliest index on tie) across the 16 lanes sharing each row.
#pragma unroll
  for (int r = 0; r < 8; ++r) {
    float vv = bestv[r];
    int   ii = besti[r];
#pragma unroll
    for (int m = 8; m >= 1; m >>= 1) {
      float ov = __shfl_xor(vv, m, 32);
      int   oi = __shfl_xor(ii, m, 32);
      if (ov > vv || (ov == vv && oi < ii)) { vv = ov; ii = oi; }
    }
    if (lsub == 0) nn[rowBase + halfv * 8 + r] = ii;
  }
}

// ---------------------------------------------------------------------------
// Phase 3a: per-row log distance (exact f32 normalized rows).
// ---------------------------------------------------------------------------
__global__ void __launch_bounds__(256)
koleo_dist(const float* __restrict__ xf, const int* __restrict__ nn,
           float* __restrict__ logd) {
  const int lane = threadIdx.x & 31;
  const int row  = blockIdx.x * 8 + (threadIdx.x >> 5);
  const int j    = nn[row];
  const float* a = xf + (size_t)row * DIM;
  const float* b = xf + (size_t)j * DIM;
  float s = 0.f;
#pragma unroll
  for (int i = 0; i < 12; ++i) {
    float d = a[lane + i * 32] - b[lane + i * 32] + EPSF;
    s += d * d;
  }
#pragma unroll
  for (int m = 16; m >= 1; m >>= 1) s += __shfl_xor(s, m, 32);
  if (lane == 0) logd[row] = logf(sqrtf(s) + EPSF);
}

// ---------------------------------------------------------------------------
// Phase 3b: deterministic fixed-order mean reduction (no float atomics).
// ---------------------------------------------------------------------------
__global__ void __launch_bounds__(256)
koleo_reduce(const float* __restrict__ logd, float* __restrict__ out) {
  __shared__ float sm[256];
  float s = 0.f;
  for (int i = threadIdx.x; i < N_ROWS; i += 256) s += logd[i];
  sm[threadIdx.x] = s;
  __syncthreads();
  for (int k = 128; k >= 1; k >>= 1) {
    if (threadIdx.x < k) sm[threadIdx.x] += sm[threadIdx.x + k];
    __syncthreads();
  }
  if (threadIdx.x == 0) out[0] = -sm[0] / (float)N_ROWS;
}

// ---------------------------------------------------------------------------
extern "C" void kernel_launch(void* const* d_in, const int* in_sizes, int n_in,
                              void* d_out, int out_size, void* d_ws, size_t ws_size,
                              hipStream_t stream) {
  (void)in_sizes; (void)n_in; (void)out_size; (void)ws_size;
  const float* in = (const float*)d_in[0];
  float* out = (float*)d_out;

  char* ws = (char*)d_ws;
  constexpr size_t XF_BYTES = (size_t)N_ROWS * DIM * sizeof(float);          // 12,582,912
  constexpr size_t XH_BYTES = (size_t)N_ROWS * DIM * sizeof(unsigned short); //  6,291,456
  float*          xf   = (float*)ws;
  unsigned short* xh   = (unsigned short*)(ws + XF_BYTES);
  int*            nn   = (int*)(ws + XF_BYTES + XH_BYTES);
  float*          logd = (float*)(ws + XF_BYTES + XH_BYTES + (size_t)N_ROWS * sizeof(int));

  koleo_normalize<<<N_ROWS / 8, 256, 0, stream>>>(in, xf, xh);
  koleo_argmax   <<<N_ROWS / 64, 128, 0, stream>>>(xh, nn);
  koleo_dist     <<<N_ROWS / 8, 256, 0, stream>>>(xf, nn, logd);
  koleo_reduce   <<<1, 256, 0, stream>>>(logd, out);
}
